// Inverted_Attention_81140522156781
// MI455X (gfx1250) — matrix operand order repaired and checked by reading, not compiled, not run on hardware
//
#include <hip/hip_runtime.h>

typedef __attribute__((ext_vector_type(16))) _Float16 v16h;
typedef __attribute__((ext_vector_type(8)))  _Float16 v8h;
typedef __attribute__((ext_vector_type(4)))  _Float16 v4h;
typedef __attribute__((ext_vector_type(2)))  _Float16 v2h;
typedef __attribute__((ext_vector_type(8)))  float    v8f;

#define NUM_HEADS 6
#define NTOK 64            // 8x8 window tokens
#define HDIM 32            // head_dim = 192/6
#define CCH  192
#define IMGW 128

// v_cvt_pk_rtz_f16_f32: two f32 -> packed 2xf16 in one VALU op
__device__ __forceinline__ v2h pkh(float x, float y) {
  auto t = __builtin_amdgcn_cvt_pkrtz(x, y);
  v2h r; __builtin_memcpy(&r, &t, sizeof(r));
  return r;
}

// ds_swizzle_b32, group-of-32 XOR mode: 1 DS op per butterfly stage, no VALU.
// offset = (xor_mask<<10) | (or_mask<<5) | and_mask ; and_mask=0x1F, xor<16
// keeps the two 16-lane halves (rows M and M+8) independent.
#define SWZF(x, imm) __int_as_float(__builtin_amdgcn_ds_swizzle(__float_as_int(x), (imm)))

__device__ __forceinline__ float redmax16(float x) {
  x = fmaxf(x, SWZF(x, 0x041F));
  x = fmaxf(x, SWZF(x, 0x081F));
  x = fmaxf(x, SWZF(x, 0x101F));
  x = fmaxf(x, SWZF(x, 0x201F));
  return x;
}
__device__ __forceinline__ float redsum16(float x) {
  x += SWZF(x, 0x041F);
  x += SWZF(x, 0x081F);
  x += SWZF(x, 0x101F);
  x += SWZF(x, 0x201F);
  return x;
}

// Convert 16 f32 (4x float4) -> 16 f16, two packed b128 LDS stores
__device__ __forceinline__ void cvt_store16(_Float16* dst, const float4* src) {
  union { v2h h[8]; v8h v[2]; } u;
  #pragma unroll
  for (int i = 0; i < 4; ++i) {
    float4 f = src[i];
    u.h[2 * i + 0] = pkh(f.x, f.y);
    u.h[2 * i + 1] = pkh(f.z, f.w);
  }
  ((v8h*)dst)[0] = u.v[0];
  ((v8h*)dst)[1] = u.v[1];
}

// One block (4 wave32) per (window, head). 6144 blocks total.
__global__ __launch_bounds__(128) void inv_attn_kernel(
    const float* __restrict__ qkv1,   // (4, B, L, C): q1,k1,v1,v2
    const float* __restrict__ qkv2,   // (2, B, L, C): q2,k2
    float* __restrict__ out,          // (B, H, W, C)
    int nB)
{
  const int bid  = blockIdx.x;
  const int hd   = bid % NUM_HEADS;
  const int w    = bid / NUM_HEADS;
  const int ww   = w & 15;            // W/8 = 16 windows per row
  const int wh   = (w >> 4) & 15;
  const int b    = w >> 8;

  const int tid   = threadIdx.x;
  const int lane  = tid & 31;
  const int wid   = tid >> 5;
  const int hlane = lane & 15;

  __shared__ __align__(32) _Float16 sQ1[NTOK * HDIM];
  __shared__ __align__(32) _Float16 sK1[NTOK * HDIM];
  __shared__ __align__(32) _Float16 sQ2[NTOK * HDIM];
  __shared__ __align__(32) _Float16 sK2[NTOK * HDIM];
  __shared__ __align__(32) _Float16 sVT[HDIM * NTOK];   // (v1+v2) transposed: [d][tok]
  __shared__ __align__(32) _Float16 sP [NTOK * NTOK];   // softmax probs, row-major

  const size_t planeSz = (size_t)nB * (IMGW * IMGW) * CCH;
  const float  scale   = 0.17677669529663687f;          // 32^-0.5

  // ---- Stage Q1,K1,Q2,K2: thread -> (token, 16-dim half); packed b128 stores ----
  {
    const int tok = tid >> 1;               // 0..63
    const int d0  = (tid & 1) * 16;         // 0 or 16
    const int rg  = wh * 8 + (tok >> 3);
    const int cg  = ww * 8 + (tok & 7);
    const size_t goff = (((size_t)b * (IMGW * IMGW) + (size_t)rg * IMGW + cg) * CCH)
                        + hd * HDIM + d0;
    const int rb = tok * HDIM + d0;
    cvt_store16(&sQ1[rb], (const float4*)(qkv1 + 0 * planeSz + goff));
    cvt_store16(&sK1[rb], (const float4*)(qkv1 + 1 * planeSz + goff));
    cvt_store16(&sQ2[rb], (const float4*)(qkv2 + 0 * planeSz + goff));
    cvt_store16(&sK2[rb], (const float4*)(qkv2 + 1 * planeSz + goff));
  }

  // ---- Stage V^T = (v1+v2)^T: thread -> (4 tokens, 4 dims); packed b64 stores ----
  // For fixed d, 4 consecutive tokens are contiguous in sVT[d][tok].
  {
    const int dg = tid & 7;                 // 8 threads span one token's 32 dims
    const int tg = tid >> 3;                // 16 groups of 4 tokens
    const int d0 = dg * 4;
    float sv[4][4];
    #pragma unroll
    for (int j = 0; j < 4; ++j) {
      const int tok = tg * 4 + j;
      const int rg  = wh * 8 + (tok >> 3);
      const int cg  = ww * 8 + (tok & 7);
      const size_t goff = (((size_t)b * (IMGW * IMGW) + (size_t)rg * IMGW + cg) * CCH)
                          + hd * HDIM + d0;
      float4 e = *(const float4*)(qkv1 + 2 * planeSz + goff);   // v1
      float4 f = *(const float4*)(qkv1 + 3 * planeSz + goff);   // v2
      sv[j][0] = e.x + f.x; sv[j][1] = e.y + f.y;
      sv[j][2] = e.z + f.z; sv[j][3] = e.w + f.w;
    }
    #pragma unroll
    for (int i = 0; i < 4; ++i) {
      union { v2h h[2]; v4h v; } u;
      u.h[0] = pkh(sv[0][i], sv[1][i]);
      u.h[1] = pkh(sv[2][i], sv[3][i]);
      *(v4h*)&sVT[(d0 + i) * NTOK + tg * 4] = u.v;
    }
  }
  __syncthreads();

  // ---- Scores: acc = q2*k1^T + k2*q1^T for one 16-row strip (wave = mtile) ----
  // A-matrix f16 16x32 lane layout: K contiguous from kB = 16*(lane>=16), row M = lane%16
  // B-matrix f16 32x16 lane layout: K contiguous from kB = 16*(lane>=16), col N = lane%16
  const int kA   = (lane >> 4) << 3;
  const int kB   = (lane >> 4) << 4;
  const int mrow = wid * 16 + hlane;

  v8h q2lo = *(const v8h*)&sQ2[mrow * HDIM + kB];
  v8h q2hi = *(const v8h*)&sQ2[mrow * HDIM + kB + 8];
  v8h k2lo = *(const v8h*)&sK2[mrow * HDIM + kB];
  v8h k2hi = *(const v8h*)&sK2[mrow * HDIM + kB + 8];
  v16h aQ2 = __builtin_shufflevector(q2lo, q2hi, 0,1,2,3,4,5,6,7,8,9,10,11,12,13,14,15);
  v16h aK2 = __builtin_shufflevector(k2lo, k2hi, 0,1,2,3,4,5,6,7,8,9,10,11,12,13,14,15);

  v8f acc[4];
  #pragma unroll
  for (int nt = 0; nt < 4; ++nt) {
    const int col = nt * 16 + hlane;
    v16h bK1 = *(const v16h*)&sK1[col * HDIM + kB];
    v16h bQ1 = *(const v16h*)&sQ1[col * HDIM + kB];
    v8f c = {};
    c = __builtin_amdgcn_wmma_f32_16x16x32_f16(false, aQ2, false, bK1, (short)0, c, false, false);
    c = __builtin_amdgcn_wmma_f32_16x16x32_f16(false, aK2, false, bQ1, (short)0, c, false, false);
    acc[nt] = c;
  }

  // ---- S = 2 - scale*acc, folded into one FMA per element ----
  // C layout: VGPR v -> row (v + 8*(lane>=16)), lane%16 -> col
  #pragma unroll
  for (int nt = 0; nt < 4; ++nt)
    #pragma unroll
    for (int v = 0; v < 8; ++v)
      acc[nt][v] = __builtin_fmaf(-scale, acc[nt][v], 2.0f);

  // ---- Row softmax: swizzle butterflies over the 16-lane halves ----
  #pragma unroll
  for (int v = 0; v < 8; ++v) {
    float rm = redmax16(fmaxf(fmaxf(acc[0][v], acc[1][v]), fmaxf(acc[2][v], acc[3][v])));
    float rs = 0.0f;
    #pragma unroll
    for (int nt = 0; nt < 4; ++nt) { acc[nt][v] = __expf(acc[nt][v] - rm); rs += acc[nt][v]; }
    rs = redsum16(rs);
    const float inv = __builtin_amdgcn_rcpf(rs);   // v_rcp_f32, not full IEEE divide
    #pragma unroll
    for (int nt = 0; nt < 4; ++nt) acc[nt][v] *= inv;
  }

  // ---- Store P (f16 row-major) for re-layout into A operand ----
  const int mbase = wid * 16 + ((lane >> 4) << 3);
  #pragma unroll
  for (int nt = 0; nt < 4; ++nt)
    #pragma unroll
    for (int v = 0; v < 8; ++v)
      sP[(mbase + v) * NTOK + nt * 16 + hlane] = (_Float16)acc[nt][v];
  __syncthreads();

  // ---- out strip = P(16x64) @ V(64x32): K=64 -> two K=32 WMMA steps ----
  v8f o0 = {}, o1 = {};
  #pragma unroll
  for (int ks = 0; ks < 2; ++ks) {
    v8h plo = *(const v8h*)&sP[mrow * NTOK + ks * 32 + kB];
    v8h phi = *(const v8h*)&sP[mrow * NTOK + ks * 32 + kB + 8];
    v16h aP = __builtin_shufflevector(plo, phi, 0,1,2,3,4,5,6,7,8,9,10,11,12,13,14,15);
    v16h bV0 = *(const v16h*)&sVT[(0 * 16 + hlane) * NTOK + ks * 32 + kB];
    v16h bV1 = *(const v16h*)&sVT[(1 * 16 + hlane) * NTOK + ks * 32 + kB];
    o0 = __builtin_amdgcn_wmma_f32_16x16x32_f16(false, aP, false, bV0, (short)0, o0, false, false);
    o1 = __builtin_amdgcn_wmma_f32_16x16x32_f16(false, aP, false, bV1, (short)0, o1, false, false);
  }

  // ---- Store: out[b][wh*8+r][ww*8+c][hd*32 + d], f32 (64B segments per half-wave) ----
  #pragma unroll
  for (int v = 0; v < 8; ++v) {
    const int M  = mbase + v;
    const int rg = wh * 8 + (M >> 3);
    const int cg = ww * 8 + (M & 7);
    const size_t obase = (((size_t)b * (IMGW * IMGW) + (size_t)rg * IMGW + cg) * CCH)
                         + hd * HDIM + hlane;
    out[obase]      = o0[v];
    out[obase + 16] = o1[v];
  }
}

extern "C" void kernel_launch(void* const* d_in, const int* in_sizes, int n_in,
                              void* d_out, int out_size, void* d_ws, size_t ws_size,
                              hipStream_t stream) {
  const float* qkv1 = (const float*)d_in[0];   // (4,B,L,C) f32
  const float* qkv2 = (const float*)d_in[1];   // (2,B,L,C) f32
  float* out = (float*)d_out;                  // (B,H,W,C) f32

  const int nB   = (int)((size_t)in_sizes[0] / (4ull * IMGW * IMGW * CCH));
  const int nWin = nB * (IMGW / 8) * (IMGW / 8);
  dim3 grid(nWin * NUM_HEADS);
  inv_attn_kernel<<<grid, 128, 0, stream>>>(qkv1, qkv2, out, nB);
}
